// SharedVectorQuantizer_26706106646575
// MI455X (gfx1250) — matrix operand order repaired and checked by reading, not compiled, not run on hardware
//
#include <hip/hip_runtime.h>
#include <hip/hip_bf16.h>
#include <float.h>

typedef _Float16 v16h __attribute__((ext_vector_type(16)));
typedef _Float16 v8h  __attribute__((ext_vector_type(8)));
typedef float    v8f  __attribute__((ext_vector_type(8)));
typedef unsigned int v4u __attribute__((ext_vector_type(4)));
typedef int      v4i  __attribute__((ext_vector_type(4)));
typedef int      v8i  __attribute__((ext_vector_type(8)));

#define M_ROWS  65536   // 16 * 4096 rows of x
#define DIM     128     // embedding dim (K of the GEMM)
#define NCODES  4096    // codebook entries (N of the GEMM)
#define NT      64      // codes per TDM tile (4 WMMA N-subtiles)
#define NTILES  (NCODES / NT)
#define ROW_PAD 136     // halfs per LDS row: 128 + 8 pad (272B stride)
#define BUF_HALFS (NT * ROW_PAD)   // 8704 halfs = 17408 B per buffer

// ---------------------------------------------------------------------------
// Kernel 1a: codebook squared norms  cnorm[k] = ||e_k||^2
// ---------------------------------------------------------------------------
__global__ void vq_cnorm_kernel(const float* __restrict__ cb,
                                float* __restrict__ cnorm) {
    const int k = blockIdx.x * blockDim.x + threadIdx.x;
    if (k >= NCODES) return;
    const float4* p = (const float4*)(cb + (size_t)k * DIM);
    float s = 0.0f;
#pragma unroll
    for (int i = 0; i < DIM / 4; ++i) {
        const float4 v = p[i];
        s += v.x * v.x + v.y * v.y + v.z * v.z + v.w * v.w;
    }
    cnorm[k] = s;
}

// ---------------------------------------------------------------------------
// Kernel 1b: one-time codebook f32 -> f16 conversion (so the GEMM staging is
// a pure TDM DMA with no VALU conversion work).
// ---------------------------------------------------------------------------
__global__ void vq_cb_f16_kernel(const float* __restrict__ cb,
                                 _Float16* __restrict__ cb16) {
    const int i = (blockIdx.x * blockDim.x + threadIdx.x) * 8;
    const float4 c0 = *(const float4*)(cb + i);
    const float4 c1 = *(const float4*)(cb + i + 4);
    v8h h;
    h[0] = (_Float16)c0.x; h[1] = (_Float16)c0.y;
    h[2] = (_Float16)c0.z; h[3] = (_Float16)c0.w;
    h[4] = (_Float16)c1.x; h[5] = (_Float16)c1.y;
    h[6] = (_Float16)c1.z; h[7] = (_Float16)c1.w;
    *(v8h*)(cb16 + i) = h;
}

// ---------------------------------------------------------------------------
// TDM issue: DMA one 64-code x 128-dim f16 tile (16 KB) from global into LDS.
// D# pad_enable inserts 4 dwords after every 64 dwords -> LDS row stride of
// 136 halfs (272 B), which rotates banks by 4 per row: conflict-free B reads.
// ---------------------------------------------------------------------------
__device__ __forceinline__ void tdm_issue_tile(const _Float16* cb16, int tile,
                                               unsigned lds_off) {
    const unsigned long long ga =
        (unsigned long long)(size_t)(cb16 + (size_t)tile * NT * DIM);
    v4u g0;
    g0[0] = 1u;                                   // count=1, load, no gather
    g0[1] = lds_off;                              // lds_addr (bytes)
    g0[2] = (unsigned)ga;                         // global_addr[31:0]
    g0[3] = (unsigned)((ga >> 32) & 0x01FFFFFFu)  // global_addr[56:32]
          | (2u << 30);                           // type=2 ("image")
    v8i g1;
    g1[0] = (1 << 16)      // data_size = 2 bytes
          | (1 << 20)      // pad_enable
          | (5 << 22)      // pad_interval: 64 dwords (2^(5+1))
          | (3 << 25);     // pad_amount: 4 dwords (3+1)
    g1[1] = (int)(128u << 16);   // tensor_dim0 = 128 (bits [79:48] low half)
    g1[2] = (int)((unsigned)NT << 16); // tensor_dim1 = 64 (bits [95:80])
    g1[3] = (int)(128u << 16);   // tile_dim0 = 128 (bits [127:112])
    g1[4] = NT;                  // tile_dim1 = 64 (bits [143:128])
    g1[5] = 128;                 // tensor_dim0_stride = 128 (bits [191:160])
    g1[6] = 0;
    g1[7] = 0;
    const v4i z4 = {0, 0, 0, 0};               // groups 2/3 unused (2-D tile)
    const v8i z8 = {0, 0, 0, 0, 0, 0, 0, 0};   // extra group (clang-23 form)
    __builtin_amdgcn_tensor_load_to_lds(g0, g1, z4, z4, z8, 0);
}

// ---------------------------------------------------------------------------
// Load one B operand (32x16 f16, ISA layout) from the padded LDS tile.
// ---------------------------------------------------------------------------
__device__ __forceinline__ v16h load_b(const _Float16* __restrict__ base,
                                       int off) {
    const v8h b0 = *(const v8h*)(base + off);
    const v8h b1 = *(const v8h*)(base + off + 16);
    v16h b;
#pragma unroll
    for (int i = 0; i < 8; ++i) { b[i] = b0[i]; b[i + 8] = b1[i]; }
    return b;
}

// ---------------------------------------------------------------------------
// Load one A tile set (16 rows x 128 K as 4 tiles of 16x32 f16) from x.
// ISA A layout (16-bit, 16x32): lanes 0-15 = rows M=0..15, halfs =
// K{0..7,16..23}; lanes 16-31 = same rows, halfs = K{8..15,24..31}.
// ---------------------------------------------------------------------------
__device__ __forceinline__ void load_a(const float* __restrict__ xp, v16h a[4]) {
#pragma unroll
    for (int kt = 0; kt < 4; ++kt) {
        const float4 f0 = *(const float4*)(xp + kt * 32 + 0);
        const float4 f1 = *(const float4*)(xp + kt * 32 + 4);
        const float4 f2 = *(const float4*)(xp + kt * 32 + 16);
        const float4 f3 = *(const float4*)(xp + kt * 32 + 20);
        a[kt][0]  = (_Float16)f0.x; a[kt][1]  = (_Float16)f0.y;
        a[kt][2]  = (_Float16)f0.z; a[kt][3]  = (_Float16)f0.w;
        a[kt][4]  = (_Float16)f1.x; a[kt][5]  = (_Float16)f1.y;
        a[kt][6]  = (_Float16)f1.z; a[kt][7]  = (_Float16)f1.w;
        a[kt][8]  = (_Float16)f2.x; a[kt][9]  = (_Float16)f2.y;
        a[kt][10] = (_Float16)f2.z; a[kt][11] = (_Float16)f2.w;
        a[kt][12] = (_Float16)f3.x; a[kt][13] = (_Float16)f3.y;
        a[kt][14] = (_Float16)f3.z; a[kt][15] = (_Float16)f3.w;
    }
}

// ---------------------------------------------------------------------------
// Kernel 2: WMMA distance GEMM + per-row argmin, TDM double-buffered staging.
// Block = 256 threads = 8 wave32 waves; each wave owns 32 rows of x (two A
// tiles sharing every B load) and per iteration computes a 32x64 score tile =
// 32 WMMAs, with B operands manually double-buffered in registers.
//   d2[m,k] = ||e_k||^2 - 2 * (x_m . e_k)
// ---------------------------------------------------------------------------
__global__ void __launch_bounds__(256)
vq_argmin_kernel(const float* __restrict__ x,
                 const _Float16* __restrict__ cb16,
                 const float* __restrict__ cnorm,
                 int*   __restrict__ tokens_i,
                 float* __restrict__ tokens_f) {
    __shared__ __align__(16) _Float16 sB[2][BUF_HALFS];

    const int tid  = threadIdx.x;
    const int wave = tid >> 5;
    const int lane = tid & 31;
    const int n    = lane & 15;     // column within 16-code subtile / A row
    const int hi   = lane >> 4;     // which half of the wave
    const int rows_base = blockIdx.x * 256 + wave * 32;
    const unsigned lds_base = (unsigned)(size_t)(void*)&sB[0][0];

    // Two A tiles: rows [rows_base, +16) and [rows_base+16, +32)
    v16h a0[4], a1[4];
    {
        const float* xp0 = x + (size_t)(rows_base + n) * DIM + (hi ? 8 : 0);
        const float* xp1 = x + (size_t)(rows_base + 16 + n) * DIM + (hi ? 8 : 0);
        load_a(xp0, a0);
        load_a(xp1, a1);
    }

    float minv[2][8];
    int   mini[2][8];
#pragma unroll
    for (int p = 0; p < 2; ++p)
#pragma unroll
        for (int r = 0; r < 8; ++r) { minv[p][r] = FLT_MAX; mini[p][r] = 0; }

    // Prologue: wave 0 kicks off the DMA of tile 0 into buffer 0.
    if (wave == 0) tdm_issue_tile(cb16, 0, lds_base);

    const int lbase = n * ROW_PAD + hi * 8;   // per-lane base within a row

    for (int t = 0; t < NTILES; ++t) {
        const int buf = t & 1;
        if (wave == 0) {
            __builtin_amdgcn_s_wait_tensorcnt(0);   // tile t landed in LDS
        }
        __syncthreads();  // publish buf; all waves finished reading buf^1
        if (wave == 0 && t + 1 < NTILES) {
            tdm_issue_tile(cb16, t + 1,
                           lds_base + (unsigned)((buf ^ 1) * BUF_HALFS * 2));
        }

        float cn[4];
#pragma unroll
        for (int sub = 0; sub < 4; ++sub)
            cn[sub] = cnorm[t * NT + sub * 16 + n];

        v8f acc0[4], acc1[4];
#pragma unroll
        for (int sub = 0; sub < 4; ++sub) {
            acc0[sub] = (v8f){0.f, 0.f, 0.f, 0.f, 0.f, 0.f, 0.f, 0.f};
            acc1[sub] = (v8f){0.f, 0.f, 0.f, 0.f, 0.f, 0.f, 0.f, 0.f};
        }

        const _Float16* base = &sB[buf][0];
        // 16 steps of (kt, sub); B register-double-buffered so the LDS load
        // for step s+1 is in flight while the two WMMAs of step s execute.
        v16h bcur = load_b(base, lbase);   // kt=0, sub=0
#pragma unroll
        for (int step = 0; step < 16; ++step) {
            const int kt  = step >> 2;
            const int sub = step & 3;
            v16h bnxt;
            if (step < 15) {
                const int kt2  = (step + 1) >> 2;
                const int sub2 = (step + 1) & 3;
                bnxt = load_b(base, sub2 * 16 * ROW_PAD + kt2 * 32 + lbase);
            }
            acc0[sub] = __builtin_amdgcn_wmma_f32_16x16x32_f16(
                false, a0[kt], false, bcur, (short)0, acc0[sub], false, false);
            acc1[sub] = __builtin_amdgcn_wmma_f32_16x16x32_f16(
                false, a1[kt], false, bcur, (short)0, acc1[sub], false, false);
            if (step < 15) bcur = bnxt;
        }

#pragma unroll
        for (int sub = 0; sub < 4; ++sub) {
            const int code = t * NT + sub * 16 + n;
#pragma unroll
            for (int r = 0; r < 8; ++r) {
                const float d0 = cn[sub] - 2.0f * acc0[sub][r];
                if (d0 < minv[0][r]) { minv[0][r] = d0; mini[0][r] = code; }
                const float d1 = cn[sub] - 2.0f * acc1[sub][r];
                if (d1 < minv[1][r]) { minv[1][r] = d1; mini[1][r] = code; }
            }
        }
        __syncthreads();  // all waves done with buf before it is re-filled
    }

    // ---- Cross-lane argmin within each 16-lane half (C layout: lanes 0-15
    // hold rows 0..7, lanes 16-31 hold rows 8..15). Tie-break: lowest index.
#pragma unroll
    for (int mask = 1; mask < 16; mask <<= 1) {
#pragma unroll
        for (int p = 0; p < 2; ++p)
#pragma unroll
        for (int r = 0; r < 8; ++r) {
            const float ov = __shfl_xor(minv[p][r], mask, 32);
            const int   oi = __shfl_xor(mini[p][r], mask, 32);
            if (ov < minv[p][r] || (ov == minv[p][r] && oi < mini[p][r])) {
                minv[p][r] = ov; mini[p][r] = oi;
            }
        }
    }
    if (n == 0) {
#pragma unroll
        for (int p = 0; p < 2; ++p)
#pragma unroll
        for (int r = 0; r < 8; ++r) {
            const int row = rows_base + p * 16 + hi * 8 + r;
            tokens_i[row] = mini[p][r];
            tokens_f[row] = (float)mini[p][r];
        }
    }
}

// ---------------------------------------------------------------------------
// Kernel 3: gather quantized vectors, STE output, per-block partial MSE sum.
// Deterministic tree reduction (no float atomics).
// ---------------------------------------------------------------------------
__global__ void __launch_bounds__(256)
vq_gather_loss_kernel(const float* __restrict__ x,
                      const float* __restrict__ cb,
                      const int*   __restrict__ tokens,
                      float* __restrict__ qout,
                      float* __restrict__ partials) {
    __shared__ float sred[256];
    const int tid = threadIdx.x;
    const int nthreads = gridDim.x * 256;
    const float4* x4 = (const float4*)x;
    const float4* c4 = (const float4*)cb;
    float4* q4 = (float4*)qout;
    const int total = M_ROWS * (DIM / 4);

    float acc = 0.0f;
    for (int u = blockIdx.x * 256 + tid; u < total; u += nthreads) {
        const int row = u >> 5;      // DIM/4 == 32 chunks per row
        const int ch  = u & 31;
        const int tok = tokens[row];
        const float4 q  = c4[tok * 32 + ch];
        const float4 xv = x4[u];
        float4 d, o;
        d.x = q.x - xv.x; d.y = q.y - xv.y; d.z = q.z - xv.z; d.w = q.w - xv.w;
        o.x = xv.x + d.x; o.y = xv.y + d.y; o.z = xv.z + d.z; o.w = xv.w + d.w;
        q4[u] = o;
        acc += d.x * d.x + d.y * d.y + d.z * d.z + d.w * d.w;
    }
    sred[tid] = acc;
    __syncthreads();
    for (int s = 128; s > 0; s >>= 1) {
        if (tid < s) sred[tid] += sred[tid + s];
        __syncthreads();
    }
    if (tid == 0) partials[blockIdx.x] = sred[0];
}

// ---------------------------------------------------------------------------
// Kernel 4: final loss = (1 + BETA) * sum / (B*N*D),  BETA = 0.5
// ---------------------------------------------------------------------------
__global__ void vq_loss_finalize_kernel(const float* __restrict__ partials,
                                        int np, float* __restrict__ loss_out) {
    __shared__ float sred[256];
    const int tid = threadIdx.x;
    float s = 0.0f;
    for (int i = tid; i < np; i += 256) s += partials[i];
    sred[tid] = s;
    __syncthreads();
    for (int k = 128; k > 0; k >>= 1) {
        if (tid < k) sred[tid] += sred[tid + k];
        __syncthreads();
    }
    if (tid == 0)
        loss_out[0] = 1.5f * sred[0] * (1.0f / ((float)M_ROWS * (float)DIM));
}

// ---------------------------------------------------------------------------
extern "C" void kernel_launch(void* const* d_in, const int* in_sizes, int n_in,
                              void* d_out, int out_size, void* d_ws, size_t ws_size,
                              hipStream_t stream) {
    const float* x  = (const float*)d_in[0];   // [16,4096,128] f32
    const float* cb = (const float*)d_in[1];   // [4096,128]    f32

    float* out      = (float*)d_out;
    float* tokens_f = out;                           // [65536]
    float* qout     = out + M_ROWS;                  // [65536*128]
    float* loss     = out + M_ROWS + M_ROWS * DIM;   // [1]

    char* ws = (char*)d_ws;
    int*      tokens_i = (int*)ws;                                   // 256 KB
    float*    cnorm    = (float*)(ws + (size_t)M_ROWS * 4);          // 16 KB
    _Float16* cb16     = (_Float16*)(ws + (size_t)M_ROWS * 4
                                        + (size_t)NCODES * 4);       // 1 MB
    float*    partials = (float*)(ws + (size_t)M_ROWS * 4
                                     + (size_t)NCODES * 4
                                     + (size_t)NCODES * DIM * 2);    // 2 KB

    vq_cnorm_kernel<<<NCODES / 256, 256, 0, stream>>>(cb, cnorm);
    vq_cb_f16_kernel<<<(NCODES * DIM) / (256 * 8), 256, 0, stream>>>(cb, cb16);
    vq_argmin_kernel<<<M_ROWS / 256, 256, 0, stream>>>(x, cb16, cnorm,
                                                       tokens_i, tokens_f);
    vq_gather_loss_kernel<<<512, 256, 0, stream>>>(x, cb, tokens_i, qout, partials);
    vq_loss_finalize_kernel<<<1, 256, 0, stream>>>(partials, 512, loss);
}
